// LocalAttention_26603027432163
// MI455X (gfx1250) — compile-verified, matrix operand order checked by reading, not run on hardware
//
#include <hip/hip_runtime.h>
#include <hip/hip_bf16.h>

typedef unsigned short u16;
typedef __attribute__((ext_vector_type(16))) __bf16 v16bf;
typedef __attribute__((ext_vector_type(8)))  float  v8f;

#define DIM       1024
#define HEADS     16
#define DIM_HEAD  64
#define WSZ       128
#define SEQ       8192
#define NWIN      (SEQ / WSZ)
#define QKV_COLS  (3 * DIM)

__device__ __forceinline__ u16 f32_to_bf16(float f) {
    unsigned int u = __float_as_uint(f);
    unsigned int r = (u + 0x7FFFu + ((u >> 16) & 1u)) >> 16;
    return (u16)r;
}

// Load one 16x32 bf16 A/B fragment for v_wmma_f32_16x16x32_bf16.
// Per ISA layout: lanes 0-15 hold K={0..7,16..23}, lanes 16-31 hold K={8..15,24..31}.
// Caller passes the lane-adjusted base (row*ld + kk + (lane>=16 ? 8 : 0)); the two
// 8-element chunks live at +0 and +16 -> two b128 loads.
__device__ __forceinline__ v16bf load_frag(const u16* base) {
    union { v16bf v; uint4 q[2]; } u;
    u.q[0] = *(const uint4*)(base);
    u.q[1] = *(const uint4*)(base + 16);
    return u.v;
}

__device__ __forceinline__ v8f wmma_bf16(v16bf a, v16bf b, v8f c) {
    return __builtin_amdgcn_wmma_f32_16x16x32_bf16(false, a, false, b, (short)0, c,
                                                   false, false);
}

// ---------------------------------------------------------------------------
// Kernel 1: LayerNorm(x) * gamma -> bf16
// ---------------------------------------------------------------------------
__global__ void __launch_bounds__(256)
ln_cast_kernel(const float* __restrict__ x, const float* __restrict__ gamma,
               u16* __restrict__ xn) {
    const int row = blockIdx.x;
    const float* xr = x + (size_t)row * DIM;
    float s = 0.f, s2 = 0.f;
    for (int c = threadIdx.x; c < DIM; c += 256) {
        float v = xr[c];
        s += v; s2 += v * v;
    }
    __shared__ float red[2][8];
    #pragma unroll
    for (int m = 16; m >= 1; m >>= 1) { s += __shfl_xor(s, m); s2 += __shfl_xor(s2, m); }
    const int wid = threadIdx.x >> 5, lane = threadIdx.x & 31;
    if (lane == 0) { red[0][wid] = s; red[1][wid] = s2; }
    __syncthreads();
    if (wid == 0) {
        s  = (lane < 8) ? red[0][lane] : 0.f;
        s2 = (lane < 8) ? red[1][lane] : 0.f;
        #pragma unroll
        for (int m = 4; m >= 1; m >>= 1) { s += __shfl_xor(s, m); s2 += __shfl_xor(s2, m); }
        if (lane == 0) { red[0][0] = s; red[1][0] = s2; }
    }
    __syncthreads();
    const float mean = red[0][0] * (1.0f / DIM);
    const float var  = red[1][0] * (1.0f / DIM) - mean * mean;
    const float inv  = rsqrtf(var + 1e-5f);
    for (int c = threadIdx.x; c < DIM; c += 256)
        xn[(size_t)row * DIM + c] = f32_to_bf16((xr[c] - mean) * inv * gamma[c]);
}

// ---------------------------------------------------------------------------
// Kernel 2: transpose + cast weights: w (K x N, f32) -> wt (N x K, bf16)
// ---------------------------------------------------------------------------
__global__ void castT_kernel(const float* __restrict__ w, u16* __restrict__ wt,
                             int K, int N) {
    int idx = blockIdx.x * blockDim.x + threadIdx.x;
    if (idx >= K * N) return;
    int k = idx / N, c = idx - k * N;
    wt[(size_t)c * K + k] = f32_to_bf16(w[idx]);
}

// ---------------------------------------------------------------------------
// Kernel 3/5: C(MxN,f32) = A(MxK,bf16 row-major) * Bt(NxK,bf16)^T [+ bias]
// Ping-pong register double-buffering, k-loop unrolled 2x (K % 64 == 0).
// Wave tile 32x64, 8 waves arranged 4(M) x 2(N) -> block tile 128x128.
// ---------------------------------------------------------------------------
__global__ void __launch_bounds__(256)
gemm_bf16_kernel(const u16* __restrict__ A, const u16* __restrict__ Bt,
                 float* __restrict__ C, const float* __restrict__ bias,
                 int Ncols, int K) {
    const int wid   = threadIdx.x >> 5;
    const int lane  = threadIdx.x & 31;
    const int lm    = lane & 15;
    const int laneK = (lane & 16) ? 8 : 0;
    const int widm  = wid & 3, widn = wid >> 2;
    const int m0 = blockIdx.x * 128 + widm * 32;
    const int n0 = blockIdx.y * 128 + widn * 64;

    const u16* ap0 = A + (size_t)(m0 + lm) * K + laneK;
    const u16* ap1 = ap0 + (size_t)16 * K;
    const u16* bp[4];
    #pragma unroll
    for (int t = 0; t < 4; ++t)
        bp[t] = Bt + (size_t)(n0 + t * 16 + lm) * K + laneK;

    v8f acc[2][4] = {};

    // Two fragment buffers, ping-pong: no register rotation copies.
    v16bf a0[2], a1[2], b[2][4];

    // Prologue: buffer 0 <- k-step 0.
    a0[0] = load_frag(ap0);
    a1[0] = load_frag(ap1);
    #pragma unroll
    for (int t = 0; t < 4; ++t) b[0][t] = load_frag(bp[t]);

    for (int kk = 0; kk < K; kk += 64) {
        // Buffer 1 <- k-step kk+32 (always in range: K % 64 == 0).
        a0[1] = load_frag(ap0 + kk + 32);
        a1[1] = load_frag(ap1 + kk + 32);
        #pragma unroll
        for (int t = 0; t < 4; ++t) b[1][t] = load_frag(bp[t] + kk + 32);

        // Compute with buffer 0 while buffer-1 loads are in flight.
        #pragma unroll
        for (int t = 0; t < 4; ++t) {
            acc[0][t] = wmma_bf16(a0[0], b[0][t], acc[0][t]);
            acc[1][t] = wmma_bf16(a1[0], b[0][t], acc[1][t]);
        }

        if (kk + 64 < K) {
            // WGP-scope prefetch ahead into the near cache (global_prefetch_b8).
            __builtin_prefetch(ap0 + kk + 128, 0, 3);
            __builtin_prefetch(bp[0] + kk + 128, 0, 3);
            // Buffer 0 <- k-step kk+64.
            a0[0] = load_frag(ap0 + kk + 64);
            a1[0] = load_frag(ap1 + kk + 64);
            #pragma unroll
            for (int t = 0; t < 4; ++t) b[0][t] = load_frag(bp[t] + kk + 64);
        }

        // Compute with buffer 1 while buffer-0 loads are in flight.
        #pragma unroll
        for (int t = 0; t < 4; ++t) {
            acc[0][t] = wmma_bf16(a0[1], b[1][t], acc[0][t]);
            acc[1][t] = wmma_bf16(a1[1], b[1][t], acc[1][t]);
        }
    }

    // C/D layout: VGPR r -> row = r + 8*(lane>=16), col = lane&15
    const int rbase = (lane & 16) ? 8 : 0;
    #pragma unroll
    for (int half = 0; half < 2; ++half) {
        const int rowb = m0 + half * 16 + rbase;
        #pragma unroll
        for (int t = 0; t < 4; ++t) {
            const int col = n0 + t * 16 + lm;
            const float bv = bias ? bias[col] : 0.f;
            #pragma unroll
            for (int r = 0; r < 8; ++r)
                C[(size_t)(rowb + r) * Ncols + col] = acc[half][t][r] + bv;
        }
    }
}

// ---------------------------------------------------------------------------
// Kernel 4: windowed attention. One block per (window, head).
// LDS: q(128x64) k(256x64) vT(64x256) p(128x256), all bf16 -> 144 KB dynamic
// (well under the 320 KB/WGP limit).
// ---------------------------------------------------------------------------
__global__ void __launch_bounds__(256)
attn_kernel(const float* __restrict__ qkv, const float* __restrict__ sinb,
            const float* __restrict__ cosb, u16* __restrict__ attn_out) {
    extern __shared__ u16 smem[];
    u16* qs  = smem;                  // 128*64
    u16* ks  = qs  + 128 * 64;        // 256*64
    u16* vts = ks  + 256 * 64;        // 64*256 (transposed: vT[d][j])
    u16* ps  = vts + 64 * 256;        // 128*256

    const int w = blockIdx.x;   // window
    const int h = blockIdx.y;   // head
    const int tid = threadIdx.x;

    // ---- Phase 1: load + RoPE + cast to LDS ----
    for (int p = tid; p < 128 * 32; p += 256) {       // q: 4096 rotation pairs
        int row = p >> 5, col = (p & 31) * 2;
        int n = w * WSZ + row;
        size_t base = (size_t)n * QKV_COLS + h * DIM_HEAD + col;
        float t0 = qkv[base], t1 = qkv[base + 1];
        float sn = sinb[n * DIM_HEAD + col], cs = cosb[n * DIM_HEAD + col];
        qs[row * 64 + col]     = f32_to_bf16(t0 * cs - t1 * sn);
        qs[row * 64 + col + 1] = f32_to_bf16(t1 * cs + t0 * sn);
    }
    for (int p = tid; p < 256 * 32; p += 256) {       // k & v: 256 rows (prev|cur)
        int row = p >> 5, col = (p & 31) * 2;
        int n = (w - 1) * WSZ + row;                  // rows>=128 land in window w
        bool valid = (row >= WSZ) || (w > 0);         // window 0: zero-padded prev
        float k0 = 0.f, k1 = 0.f, v0 = 0.f, v1 = 0.f;
        if (valid) {
            size_t base = (size_t)n * QKV_COLS + h * DIM_HEAD + col;
            float a0 = qkv[base + DIM],     a1 = qkv[base + DIM + 1];
            float b0 = qkv[base + 2 * DIM], b1 = qkv[base + 2 * DIM + 1];
            float sn = sinb[n * DIM_HEAD + col], cs = cosb[n * DIM_HEAD + col];
            k0 = a0 * cs - a1 * sn; k1 = a1 * cs + a0 * sn;
            v0 = b0 * cs - b1 * sn; v1 = b1 * cs + b0 * sn;
        }
        ks[row * 64 + col]       = f32_to_bf16(k0);
        ks[row * 64 + col + 1]   = f32_to_bf16(k1);
        vts[col * 256 + row]       = f32_to_bf16(v0);
        vts[(col + 1) * 256 + row] = f32_to_bf16(v1);
    }
    __syncthreads();

    // ---- Phase 2: S = q k^T (16x256 strip per wave), mask, softmax -> P ----
    const int wid  = tid >> 5;
    const int lane = tid & 31;
    const int lm   = lane & 15;
    const int laneK = (lane & 16) ? 8 : 0;
    const int i0 = wid * 16;
    const int rbase = (lane & 16) ? 8 : 0;

    const u16* qbase = qs + (i0 + lm) * 64 + laneK;
    v16bf a0 = load_frag(qbase);        // K = 0..31
    v16bf a1 = load_frag(qbase + 32);   // K = 32..63

    v8f s[16];
    #pragma unroll
    for (int t = 0; t < 16; ++t) {
        const u16* kb = ks + (t * 16 + lm) * 64 + laneK;
        v16bf b0 = load_frag(kb);
        v16bf b1 = load_frag(kb + 32);
        v8f acc = {};
        acc = wmma_bf16(a0, b0, acc);
        acc = wmma_bf16(a1, b1, acc);
        s[t] = acc;
    }

    const float scale = 0.125f;  // 64^-0.5
    float rmax[8], rsum[8];
    #pragma unroll
    for (int r = 0; r < 8; ++r) rmax[r] = -3.4e38f;
    #pragma unroll
    for (int t = 0; t < 16; ++t) {
        const int j = t * 16 + lm;
        #pragma unroll
        for (int r = 0; r < 8; ++r) {
            const int i = i0 + rbase + r;
            float v = s[t][r] * scale;
            if (j > i + WSZ) v = -1e10f;   // tril(..., +WSZ) mask
            s[t][r] = v;
            rmax[r] = fmaxf(rmax[r], v);
        }
    }
    #pragma unroll
    for (int r = 0; r < 8; ++r) {          // reduce across the 16 lanes of a row
        float m = rmax[r];
        #pragma unroll
        for (int msk = 8; msk >= 1; msk >>= 1) m = fmaxf(m, __shfl_xor(m, msk));
        rmax[r] = m; rsum[r] = 0.f;
    }
    #pragma unroll
    for (int t = 0; t < 16; ++t)
        #pragma unroll
        for (int r = 0; r < 8; ++r) {
            float e = __expf(s[t][r] - rmax[r]);
            s[t][r] = e;
            rsum[r] += e;
        }
    #pragma unroll
    for (int r = 0; r < 8; ++r) {
        float m = rsum[r];
        #pragma unroll
        for (int msk = 8; msk >= 1; msk >>= 1) m += __shfl_xor(m, msk);
        rsum[r] = 1.f / m;
    }
    #pragma unroll
    for (int t = 0; t < 16; ++t)
        #pragma unroll
        for (int r = 0; r < 8; ++r)
            ps[(i0 + rbase + r) * 256 + t * 16 + lm] = f32_to_bf16(s[t][r] * rsum[r]);
    __syncthreads();

    // ---- Phase 3: O = P @ V (16x64 strip per wave, K=256) ----
    v8f o[4] = {};
    const u16* pbase = ps + (i0 + lm) * 256 + laneK;
    #pragma unroll
    for (int kk = 0; kk < 256; kk += 32) {
        v16bf a = load_frag(pbase + kk);
        #pragma unroll
        for (int t = 0; t < 4; ++t) {
            v16bf b = load_frag(vts + (t * 16 + lm) * 256 + laneK + kk);
            o[t] = wmma_bf16(a, b, o[t]);
        }
    }
    #pragma unroll
    for (int t = 0; t < 4; ++t) {
        const int d = t * 16 + lm;
        #pragma unroll
        for (int r = 0; r < 8; ++r) {
            const int n = w * WSZ + i0 + rbase + r;
            attn_out[(size_t)n * DIM + h * DIM_HEAD + d] = f32_to_bf16(o[t][r]);
        }
    }
}

// ---------------------------------------------------------------------------
extern "C" void kernel_launch(void* const* d_in, const int* in_sizes, int n_in,
                              void* d_out, int out_size, void* d_ws, size_t ws_size,
                              hipStream_t stream) {
    const float* x     = (const float*)d_in[0];
    const float* sinb  = (const float*)d_in[1];
    const float* cosb  = (const float*)d_in[2];
    const float* gamma = (const float*)d_in[3];
    const float* w_qkv = (const float*)d_in[4];
    const float* w_out = (const float*)d_in[5];
    const float* b_out = (const float*)d_in[6];
    float* out = (float*)d_out;

    char* ws = (char*)d_ws;
    u16*   xn    = (u16*)(ws);                                   // 16 MB
    u16*   wqkvT = (u16*)(ws + (size_t)16  * 1024 * 1024);       //  6 MB
    u16*   woutT = (u16*)(ws + (size_t)22  * 1024 * 1024);       //  2 MB
    float* qkv   = (float*)(ws + (size_t)24 * 1024 * 1024);      // 96 MB
    u16*   attn  = (u16*)(ws + (size_t)120 * 1024 * 1024);       // 16 MB

    ln_cast_kernel<<<SEQ, 256, 0, stream>>>(x, gamma, xn);
    castT_kernel<<<(DIM * QKV_COLS + 255) / 256, 256, 0, stream>>>(w_qkv, wqkvT, DIM, QKV_COLS);
    castT_kernel<<<(DIM * DIM + 255) / 256, 256, 0, stream>>>(w_out, woutT, DIM, DIM);

    dim3 g1(SEQ / 128, QKV_COLS / 128);
    gemm_bf16_kernel<<<g1, 256, 0, stream>>>(xn, wqkvT, qkv, nullptr, QKV_COLS, DIM);

    dim3 ga(NWIN, HEADS);
    size_t smem = (size_t)(128 * 64 + 256 * 64 + 64 * 256 + 128 * 256) * sizeof(u16); // 144 KB
    attn_kernel<<<ga, 256, smem, stream>>>(qkv, sinb, cosb, attn);

    dim3 g2(SEQ / 128, DIM / 128);
    gemm_bf16_kernel<<<g2, 256, 0, stream>>>(attn, woutT, out, b_out, DIM, DIM);
}
